// AASIST_LARGE_24240795419408
// MI455X (gfx1250) — matrix operand order repaired and checked by reading, not compiled, not run on hardware
//
#include <hip/hip_runtime.h>
#include <hip/hip_bf16.h>

typedef __attribute__((ext_vector_type(16))) _Float16 v16h;
typedef __attribute__((ext_vector_type(8)))  float    v8f;

union FragH { v16h v; _Float16 h[16]; uint4 q[2]; };

// ---------------------------------------------------------------------------
// Generic WMMA GEMM: C[M x N](f32) = A[M x K](f16, row-major) * Bt[N x K]^T
// Requires M%128==0, N%128==0, K%32==0 (true for every call site here).
// Block = 256 threads = 8 waves arranged 4(M) x 2(N); each wave computes a
// 32x64 sub-tile as 2x4 grid of 16x16x32 WMMA ops.
// ---------------------------------------------------------------------------
#define TM 128
#define TN 128
#define TK 32

__global__ __launch_bounds__(256) void k_gemm_f16f32(
    const _Float16* __restrict__ A, const _Float16* __restrict__ Bt,
    float* __restrict__ C, int M, int N, int K)
{
  __shared__ __align__(16) _Float16 sA[TM][TK];
  __shared__ __align__(16) _Float16 sB[TN][TK];

  const int tid  = threadIdx.x;
  const int lane = tid & 31;
  const int wave = tid >> 5;
  const int wm   = (wave >> 1) * 32;   // wave row offset in tile
  const int wn   = (wave & 1) * 64;    // wave col offset in tile
  const int lh   = lane >> 4;          // lane half (0/1)
  const int l15  = lane & 15;

  const long bm = (long)blockIdx.y * TM;
  const long bn = (long)blockIdx.x * TN;

  v8f acc[2][4] = {};

  const int r  = tid >> 2;             // 0..63 (row within tile, two passes)
  const int cc = (tid & 3) * 8;        // 0,8,16,24 (K offset, 8 halves = 16B)

  for (int k0 = 0; k0 < K; k0 += TK) {
    // --- stage 128x32 f16 tiles of A and Bt into LDS (16B vector loads) ---
    *(uint4*)&sA[r][cc]      = *(const uint4*)&A[(bm + r) * (long)K + k0 + cc];
    *(uint4*)&sA[r + 64][cc] = *(const uint4*)&A[(bm + r + 64) * (long)K + k0 + cc];
    *(uint4*)&sB[r][cc]      = *(const uint4*)&Bt[(bn + r) * (long)K + k0 + cc];
    *(uint4*)&sB[r + 64][cc] = *(const uint4*)&Bt[(bn + r + 64) * (long)K + k0 + cc];
    if (k0 + TK < K) {   // pull next K-tile toward the caches (global_prefetch_b8)
      __builtin_prefetch(&A[(bm + r) * (long)K + k0 + TK + cc], 0, 1);
      __builtin_prefetch(&Bt[(bn + r) * (long)K + k0 + TK + cc], 0, 1);
    }
    __syncthreads();

    // --- build fragments per ISA 16-bit layouts ---
    // A 16x32: lanes 0-15 row M=l, K 0..7 & 16..23; lanes 16-31 K 8..15 & 24..31
    FragH af[2], bf[4];
    #pragma unroll
    for (int mi = 0; mi < 2; ++mi) {
      const _Float16* row = &sA[wm + mi * 16 + l15][0];
      af[mi].q[0] = *(const uint4*)&row[lh * 8];
      af[mi].q[1] = *(const uint4*)&row[16 + lh * 8];
    }
    // B 32x16: lanes 0-15 col N=l, K 0..7 & 16..23; lanes 16-31 K 8..15 & 24..31
    #pragma unroll
    for (int ni = 0; ni < 4; ++ni) {
      const _Float16* row = &sB[wn + ni * 16 + l15][0];
      bf[ni].q[0] = *(const uint4*)&row[lh * 8];
      bf[ni].q[1] = *(const uint4*)&row[16 + lh * 8];
    }

    #pragma unroll
    for (int mi = 0; mi < 2; ++mi)
      #pragma unroll
      for (int ni = 0; ni < 4; ++ni)
        acc[mi][ni] = __builtin_amdgcn_wmma_f32_16x16x32_f16(
            false, af[mi].v, false, bf[ni].v, (short)0, acc[mi][ni], false, false);
    __syncthreads();
  }

  // C/D layout: VGPR rr -> row = base + lh*8 + rr, col = l15
  #pragma unroll
  for (int mi = 0; mi < 2; ++mi)
    #pragma unroll
    for (int ni = 0; ni < 4; ++ni) {
      long row = bm + wm + mi * 16 + lh * 8;
      long col = bn + wn + ni * 16 + l15;
      #pragma unroll
      for (int rr = 0; rr < 8; ++rr)
        C[(row + rr) * (long)N + col] = acc[mi][ni][rr];
    }
}

// ---------------------------------------------------------------------------
// conv_time (Cin=1, k=3, pad=1) + BN0 + ReLU -> h0 f16 [4][4096][128]
// ---------------------------------------------------------------------------
__global__ void k_conv_time(const float* __restrict__ x,
                            const float* __restrict__ w, const float* __restrict__ bias,
                            const float* __restrict__ g, const float* __restrict__ bta,
                            const float* __restrict__ m, const float* __restrict__ v,
                            _Float16* __restrict__ out)
{
  int idx = blockIdx.x * 256 + threadIdx.x;       // 4*4096*128 = 2^21
  int c = idx & 127;
  int t = (idx >> 7) & 4095;
  int b = idx >> 19;
  const float* xb = x + (long)b * 4096;
  float acc = bias[c];
  if (t > 0)    acc += xb[t - 1] * w[c * 3 + 0];
  acc += xb[t] * w[c * 3 + 1];
  if (t < 4095) acc += xb[t + 1] * w[c * 3 + 2];
  float s = g[c] * rsqrtf(v[c] + 1e-5f);
  float y = (acc - m[c]) * s + bta[c];
  out[idx] = (_Float16)fmaxf(y, 0.0f);
}

// im2col for k=3 pad=1 conv: in [B][T][C] f16 -> A [B*T][3*C] f16 (K = dt*C + c)
__global__ void k_im2col(const _Float16* __restrict__ in, _Float16* __restrict__ outA,
                         int B, int T, int C)
{
  long idx = (long)blockIdx.x * 256 + threadIdx.x;
  long total = (long)B * T * 3 * C;
  if (idx >= total) return;
  int  c  = (int)(idx % C);
  int  dt = (int)((idx / C) % 3);
  long bt = idx / (3L * C);
  int  t  = (int)(bt % T);
  int  b  = (int)(bt / T);
  int  tt = t + dt - 1;
  _Float16 val = (_Float16)0.0f;
  if (tt >= 0 && tt < T) val = in[((long)b * T + tt) * C + c];
  outA[idx] = val;
}

// conv weight [O][C][3] f32 -> Wt [O][3*C] f16 with K = dt*C + c
__global__ void k_repack_w(const float* __restrict__ w, _Float16* __restrict__ wt,
                           int O, int C)
{
  long idx = (long)blockIdx.x * 256 + threadIdx.x;
  long total = (long)O * 3 * C;
  if (idx >= total) return;
  int c  = (int)(idx % C);
  int dt = (int)((idx / C) % 3);
  long o = idx / (3L * C);
  wt[idx] = (_Float16)w[(o * C + c) * 3 + dt];
}

// GEMM out [B*T][O] f32 -> +convbias, BN, ReLU, maxpool2 over t -> [B][T/2][O] f16
__global__ void k_bnrelu_pool(const float* __restrict__ G,
                              const float* __restrict__ bias, const float* __restrict__ gg,
                              const float* __restrict__ bb, const float* __restrict__ mm,
                              const float* __restrict__ vv,
                              _Float16* __restrict__ out, int B, int T, int O)
{
  long idx = (long)blockIdx.x * 256 + threadIdx.x;
  long total = (long)B * (T / 2) * O;
  if (idx >= total) return;
  int  o  = (int)(idx % O);
  long rr = idx / O;
  int  t2 = (int)(rr % (T / 2));
  int  b  = (int)(rr / (T / 2));
  float s = gg[o] * rsqrtf(vv[o] + 1e-5f);
  long row = (long)b * T + 2 * t2;
  float v0 = (G[row * O + o] + bias[o] - mm[o]) * s + bb[o];
  float v1 = (G[(row + 1) * O + o] + bias[o] - mm[o]) * s + bb[o];
  v0 = fmaxf(v0, 0.0f); v1 = fmaxf(v1, 0.0f);
  out[idx] = (_Float16)fmaxf(v0, v1);
}

__global__ void k_cvt(const float* __restrict__ in, _Float16* __restrict__ out, long n)
{
  long idx = (long)blockIdx.x * 256 + threadIdx.x;
  if (idx < n) out[idx] = (_Float16)in[idx];
}

// f32 [N][D] (+optional bias) -> f16 row-major and optional f16 transpose [D][N]
__global__ void k_bias_dual(const float* __restrict__ G, const float* __restrict__ bias,
                            _Float16* __restrict__ out, _Float16* __restrict__ outT,
                            int Nrows, int D)
{
  long idx = (long)blockIdx.x * 256 + threadIdx.x;
  long total = (long)Nrows * D;
  if (idx >= total) return;
  int  d = (int)(idx % D);
  long n = idx / D;
  float v = G[idx] + (bias ? bias[d] : 0.0f);
  out[idx] = (_Float16)v;
  if (outT) outT[(long)d * Nrows + n] = (_Float16)v;
}

// s1[n] = h[n,:]@aw[:D], s2[n] = h[n,:]@aw[D:]
__global__ void k_scores(const _Float16* __restrict__ h, const float* __restrict__ aw,
                         float* __restrict__ s1, float* __restrict__ s2, int N, int D)
{
  int n = blockIdx.x * 256 + threadIdx.x;
  if (n >= N) return;
  float a = 0.0f, b = 0.0f;
  const _Float16* hr = h + (long)n * D;
  for (int d = 0; d < D; ++d) {
    float hv = (float)hr[d];
    a += hv * aw[d];
    b += hv * aw[D + d];
  }
  s1[n] = a; s2[n] = b;
}

// att[i,:] = softmax_j(leaky_relu(s1[i]+s2[j]+ab)), stored f16. One block per row.
__global__ __launch_bounds__(256) void k_softmax_att(const float* __restrict__ s1,
                                                     const float* __restrict__ s2,
                                                     const float* __restrict__ abp,
                                                     _Float16* __restrict__ att, int N)
{
  __shared__ float red[256];
  int i = blockIdx.x, tid = threadIdx.x;
  float si = s1[i] + abp[0];
  float mx = -3.4e38f;
  for (int j = tid; j < N; j += 256) {
    float e = si + s2[j];
    e = e > 0.0f ? e : 0.01f * e;
    mx = fmaxf(mx, e);
  }
  red[tid] = mx; __syncthreads();
  for (int s = 128; s > 0; s >>= 1) {
    if (tid < s) red[tid] = fmaxf(red[tid], red[tid + s]);
    __syncthreads();
  }
  mx = red[0]; __syncthreads();
  float sum = 0.0f;
  for (int j = tid; j < N; j += 256) {
    float e = si + s2[j];
    e = e > 0.0f ? e : 0.01f * e;
    sum += __expf(e - mx);
  }
  red[tid] = sum; __syncthreads();
  for (int s = 128; s > 0; s >>= 1) {
    if (tid < s) red[tid] += red[tid + s];
    __syncthreads();
  }
  float inv = 1.0f / red[0];
  for (int j = tid; j < N; j += 256) {
    float e = si + s2[j];
    e = e > 0.0f ? e : 0.01f * e;
    att[(long)i * N + j] = (_Float16)(__expf(e - mx) * inv);
  }
}

// out2 [4096][256] f32 -> mean over 1024 nodes/batch -> head Linear(256,2)
__global__ __launch_bounds__(256) void k_mean_head(const float* __restrict__ G,
                                                   const float* __restrict__ fcw,
                                                   const float* __restrict__ fcb,
                                                   float* __restrict__ out)
{
  __shared__ float pooled[256];
  int b = blockIdx.x, c = threadIdx.x;
  float s = 0.0f;
  for (int t = 0; t < 1024; ++t) s += G[((long)b * 1024 + t) * 256 + c];
  pooled[c] = s * (1.0f / 1024.0f);
  __syncthreads();
  if (c < 2) {
    float acc = fcb[c];
    for (int k = 0; k < 256; ++k) acc += pooled[k] * fcw[c * 256 + k];
    out[b * 2 + c] = acc;
  }
}

// ---------------------------------------------------------------------------
extern "C" void kernel_launch(void* const* d_in, const int* in_sizes, int n_in,
                              void* d_out, int out_size, void* d_ws, size_t ws_size,
                              hipStream_t stream)
{
  const float* x      = (const float*)d_in[0];
  const float* ctw    = (const float*)d_in[1];
  const float* ctb    = (const float*)d_in[2];
  const float* bn0g   = (const float*)d_in[3];
  const float* bn0b   = (const float*)d_in[4];
  const float* bn0m   = (const float*)d_in[5];
  const float* bn0v   = (const float*)d_in[6];
  const float* c1w    = (const float*)d_in[7];
  const float* c1b    = (const float*)d_in[8];
  const float* bn1g   = (const float*)d_in[9];
  const float* bn1b   = (const float*)d_in[10];
  const float* bn1m   = (const float*)d_in[11];
  const float* bn1v   = (const float*)d_in[12];
  const float* c2w    = (const float*)d_in[13];
  const float* c2b    = (const float*)d_in[14];
  const float* bn2g   = (const float*)d_in[15];
  const float* bn2b   = (const float*)d_in[16];
  const float* bn2m   = (const float*)d_in[17];
  const float* bn2v   = (const float*)d_in[18];
  const float* g1fcw  = (const float*)d_in[19];
  const float* g1fcb  = (const float*)d_in[20];
  const float* g1aw   = (const float*)d_in[21];
  const float* g1ab   = (const float*)d_in[22];
  const float* g2fcw  = (const float*)d_in[23];
  const float* g2fcb  = (const float*)d_in[24];
  const float* g2aw   = (const float*)d_in[25];
  const float* g2ab   = (const float*)d_in[26];
  const float* fcw    = (const float*)d_in[27];
  const float* fcb    = (const float*)d_in[28];

  // workspace layout (bytes); big buffers strictly sequentially reused
  char* ws = (char*)d_ws;
  _Float16* h0    = (_Float16*)(ws + 0);          // 4*4096*128 f16       = 4,194,304 B
  _Float16* im2c  = (_Float16*)(ws + 4194304);    // max im2col 16384*384 = 12,582,912
  _Float16* wrep  = (_Float16*)(ws + 16777216);   // weight f16 scratch   =    786,432
  float*    gout  = (float*)   (ws + 17563648);   // max GEMM out f32     = 16,777,216
  _Float16* h1    = (_Float16*)(ws + 34340864);   // 4*2048*256 f16       = 4,194,304
  _Float16* gnod  = (_Float16*)(ws + 38535168);   // nodes 4096*512 f16   = 4,194,304
  _Float16* hf    = (_Float16*)(ws + 42729472);   // h / h2 f16           = 4,194,304
  _Float16* ht    = (_Float16*)(ws + 46923776);   // hT / h2T f16         = 4,194,304
  _Float16* gg2   = (_Float16*)(ws + 51118080);   // g2 nodes f16         = 4,194,304
  float*    s1    = (float*)   (ws + 55312384);   // 4096 f32
  float*    s2    = (float*)   (ws + 55328768);   // 4096 f32
  _Float16* att   = (_Float16*)(ws + 55345152);   // 4096*4096 f16        = 33,554,432
  (void)ws_size; (void)in_sizes; (void)n_in; (void)out_size;

  // ---- conv stack ----
  k_conv_time<<<8192, 256, 0, stream>>>(x, ctw, ctb, bn0g, bn0b, bn0m, bn0v, h0);
  k_im2col<<<(4 * 4096 * 3 * 128) / 256, 256, 0, stream>>>(h0, im2c, 4, 4096, 128);
  k_repack_w<<<(256 * 3 * 128 + 255) / 256, 256, 0, stream>>>(c1w, wrep, 256, 128);
  k_gemm_f16f32<<<dim3(256 / TN * 1 + (256 % TN ? 1 : 0), 16384 / TM), 256, 0, stream>>>(
      im2c, wrep, gout, 16384, 256, 384);
  k_bnrelu_pool<<<(4 * 2048 * 256) / 256, 256, 0, stream>>>(
      gout, c1b, bn1g, bn1b, bn1m, bn1v, h1, 4, 4096, 256);
  k_im2col<<<(4 * 2048 * 3 * 256) / 256, 256, 0, stream>>>(h1, im2c, 4, 2048, 256);
  k_repack_w<<<(512 * 3 * 256 + 255) / 256, 256, 0, stream>>>(c2w, wrep, 512, 256);
  k_gemm_f16f32<<<dim3(512 / TN, 8192 / TM), 256, 0, stream>>>(
      im2c, wrep, gout, 8192, 512, 768);
  k_bnrelu_pool<<<(4 * 1024 * 512) / 256, 256, 0, stream>>>(
      gout, c2b, bn2g, bn2b, bn2m, bn2v, gnod, 4, 2048, 512);

  // ---- GAT 1 ----
  k_cvt<<<(512 * 512) / 256, 256, 0, stream>>>(g1fcw, wrep, 512L * 512);
  k_gemm_f16f32<<<dim3(512 / TN, 4096 / TM), 256, 0, stream>>>(
      gnod, wrep, gout, 4096, 512, 512);
  k_bias_dual<<<(4096 * 512) / 256, 256, 0, stream>>>(gout, g1fcb, hf, ht, 4096, 512);
  k_scores<<<16, 256, 0, stream>>>(hf, g1aw, s1, s2, 4096, 512);
  k_softmax_att<<<4096, 256, 0, stream>>>(s1, s2, g1ab, att, 4096);
  k_gemm_f16f32<<<dim3(512 / TN, 4096 / TM), 256, 0, stream>>>(
      att, ht, gout, 4096, 512, 4096);
  k_bias_dual<<<(4096 * 512) / 256, 256, 0, stream>>>(gout, nullptr, gg2, nullptr, 4096, 512);

  // ---- GAT 2 ----
  k_cvt<<<(256 * 512) / 256, 256, 0, stream>>>(g2fcw, wrep, 256L * 512);
  k_gemm_f16f32<<<dim3(256 / TN, 4096 / TM), 256, 0, stream>>>(
      gg2, wrep, gout, 4096, 256, 512);
  k_bias_dual<<<(4096 * 256) / 256, 256, 0, stream>>>(gout, g2fcb, hf, ht, 4096, 256);
  k_scores<<<16, 256, 0, stream>>>(hf, g2aw, s1, s2, 4096, 256);
  k_softmax_att<<<4096, 256, 0, stream>>>(s1, s2, g2ab, att, 4096);
  k_gemm_f16f32<<<dim3(256 / TN, 4096 / TM), 256, 0, stream>>>(
      att, ht, gout, 4096, 256, 4096);

  // ---- mean pool + head ----
  k_mean_head<<<4, 256, 0, stream>>>(gout, fcw, fcb, (float*)d_out);
}
